// ModalFusion_77807627534884
// MI455X (gfx1250) — matrix operand order repaired and checked by reading, not compiled, not run on hardware
//
#include <hip/hip_runtime.h>
#include <hip/hip_bf16.h>

typedef __attribute__((ext_vector_type(16))) _Float16 v16h;
typedef __attribute__((ext_vector_type(8)))  _Float16 v8h;
typedef __attribute__((ext_vector_type(4)))  _Float16 v4h;
typedef __attribute__((ext_vector_type(8)))  float    v8f;

#define B_  32
#define N_  1024
#define M_  576
#define D_  768
#define BM  32      // Q rows per workgroup
#define MC  64      // image rows per K/V chunk
#define NCHUNK (M_ / MC)   // 9

// Padded LDS strides (elements) to avoid bank conflicts
#define QLD (D_ + 8)    // 776, q & kv row stride (f16)
#define MCP (MC + 8)    // 72,  kvT row stride (f16)
#define SLD (MC + 1)    // 65,  s row stride (f32)
#define PLD (MC + 8)    // 72,  p row stride (f16)

// LDS layout (bytes)
#define OFF_Q   0
#define OFF_KV  (OFF_Q  + BM * QLD * 2)     // 49664
#define OFF_KVT (OFF_KV + MC * QLD * 2)     // 148992
#define OFF_S   (OFF_KVT + D_ * MCP * 2)    // 259584
#define OFF_P   (OFF_S  + BM * SLD * 4)     // 267904
#define OFF_SC  (OFF_P  + BM * PLD * 2)     // 272512
#define LDS_BYTES (OFF_SC + BM * 4)         // 272640

// A fragment (16x32 f16): lane L holds row r0+(L&15); K-slices {kb..kb+7, kb+16..kb+23}, kb = (L>>4)*8
__device__ __forceinline__ v16h frag_a(const _Float16* m, int ld, int r0, int k0, int lane) {
    const _Float16* ptr = m + (r0 + (lane & 15)) * ld + k0 + ((lane >> 4) << 3);
    v8h lo = *(const v8h*)ptr;
    v8h hi = *(const v8h*)(ptr + 16);
    v16h r;
#pragma unroll
    for (int i = 0; i < 8; ++i) { r[i] = lo[i]; r[i + 8] = hi[i]; }
    return r;
}

// B fragment (32x16 f16): lane L holds column n0+(L&15); K-slices {kb..kb+7, kb+16..kb+23}, kb = k0 + (L>>4)*8
__device__ __forceinline__ v16h frag_b(const _Float16* m, int ld, int n0, int k0, int lane) {
    const _Float16* ptr = m + (n0 + (lane & 15)) * ld + k0 + ((lane >> 4) << 3);
    v8h lo = *(const v8h*)ptr;
    v8h hi = *(const v8h*)(ptr + 16);
    v16h r;
#pragma unroll
    for (int i = 0; i < 8; ++i) { r[i] = lo[i]; r[i + 8] = hi[i]; }
    return r;
}

__global__ __launch_bounds__(256) void ModalFusion_fa_kernel(
        const float* __restrict__ hs,    // [B, N, D] fp32
        const float* __restrict__ img,   // [B, M, D] fp32
        float* __restrict__ out)         // [B, N, D] fp32
{
    extern __shared__ char smem[];
    _Float16* q    = (_Float16*)(smem + OFF_Q);    // [BM][QLD]
    _Float16* kv   = (_Float16*)(smem + OFF_KV);   // [MC][QLD]  row-major chunk
    _Float16* kvT  = (_Float16*)(smem + OFF_KVT);  // [D_][MCP]  transposed chunk
    float*    sbuf = (float*)   (smem + OFF_S);    // [BM][SLD]
    _Float16* pbuf = (_Float16*)(smem + OFF_P);    // [BM][PLD]
    float*    rsc  = (float*)   (smem + OFF_SC);   // [BM] per-row scale

    const int tid  = threadIdx.x;
    const int lane = tid & 31;
    const int wave = tid >> 5;
    const int b     = blockIdx.x >> 5;   // 32 Q-tiles per batch
    const int n0    = (blockIdx.x & 31) * BM;

    const int rt = wave >> 2;            // row tile (0..1) for both S and PV phases
    const int ct = wave & 3;             // S phase: column tile; PV phase: column group

    // Softmax ownership: 8 lanes per row, replicated running stats
    const int srow = tid >> 3;           // 0..31
    const int ssub = tid & 7;            // 0..7 -> columns ssub + 8k

    // ---- Stage Q tile: fp32 -> f16 into LDS (rows are contiguous in global) ----
    {
        const float4* src = (const float4*)(hs + ((size_t)b * N_ + n0) * D_);
        const int nvec = BM * D_ / 4;                       // 6144
        for (int i = tid; i < nvec; i += 256) {
            float4 f = src[i];
            const int r = i / (D_ / 4);
            const int c = (i - r * (D_ / 4)) * 4;
            v4h h; h[0] = (_Float16)f.x; h[1] = (_Float16)f.y;
                   h[2] = (_Float16)f.z; h[3] = (_Float16)f.w;
            *(v4h*)(q + r * QLD + c) = h;
        }
    }

    // Output accumulators: 12 tiles of 16x16 f32 per wave (rows rt*16.., cols ct*192..)
    v8f acc[12];
#pragma unroll
    for (int t = 0; t < 12; ++t)
#pragma unroll
        for (int v = 0; v < 8; ++v) acc[t][v] = 0.0f;

    // Online-softmax running stats (replicated across the 8 lanes of each row group)
    float m_run = -INFINITY;
    float l_run = 0.0f;

    for (int mc = 0; mc < NCHUNK; ++mc) {
        __syncthreads();   // previous iteration done reading kv/kvT/pbuf

        // ---- Stage K/V chunk: fp32 -> f16, row-major + transposed ----
        {
            const float4* src = (const float4*)(img + ((size_t)b * M_ + mc * MC) * D_);
            const int nvec = MC * D_ / 4;                   // 12288
            for (int i = tid; i < nvec; i += 256) {
                float4 f = src[i];
                const int r = i / (D_ / 4);
                const int c = (i - r * (D_ / 4)) * 4;
                v4h h; h[0] = (_Float16)f.x; h[1] = (_Float16)f.y;
                       h[2] = (_Float16)f.z; h[3] = (_Float16)f.w;
                *(v4h*)(kv + r * QLD + c) = h;
#pragma unroll
                for (int j = 0; j < 4; ++j) kvT[(c + j) * MCP + r] = h[j];
            }
        }

        // ---- Prefetch next K/V chunk into L2 while this one is consumed ----
        if (mc + 1 < NCHUNK) {
            const char* nxt = (const char*)(img + ((size_t)b * M_ + (mc + 1) * MC) * D_);
#pragma unroll
            for (int i = 0; i < 6; ++i)   // 256 thr * 6 * 128B = full 192KB chunk
                __builtin_prefetch(nxt + ((size_t)tid + i * 256) * 128, 0, 3);
        }
        __syncthreads();

        // ---- S = Q * K^T : one 16x16 tile per wave (32x64 total) ----
        {
            v8f sacc;
#pragma unroll
            for (int v = 0; v < 8; ++v) sacc[v] = 0.0f;
#pragma unroll
            for (int kk = 0; kk < D_ / 32; ++kk) {          // 24 WMMA
                v16h a  = frag_a(q,  QLD, rt * 16, kk * 32, lane);
                v16h bt = frag_b(kv, QLD, ct * 16, kk * 32, lane);
                sacc = __builtin_amdgcn_wmma_f32_16x16x32_f16(
                        false, a, false, bt, (short)0, sacc, false, false);
            }
            const int col   = ct * 16 + (lane & 15);
            const int rbase = rt * 16 + ((lane >> 4) << 3);
#pragma unroll
            for (int v = 0; v < 8; ++v) sbuf[(rbase + v) * SLD + col] = sacc[v];
        }
        __syncthreads();

        // ---- Online softmax: 8 lanes per row, butterfly reductions ----
        {
            float sv[8];
#pragma unroll
            for (int k = 0; k < 8; ++k) sv[k] = sbuf[srow * SLD + ssub + 8 * k];
            float cmax = -INFINITY;
#pragma unroll
            for (int k = 0; k < 8; ++k) cmax = fmaxf(cmax, sv[k]);
#pragma unroll
            for (int msk = 1; msk < 8; msk <<= 1)
                cmax = fmaxf(cmax, __shfl_xor(cmax, msk, 32));
            const float m_new = fmaxf(m_run, cmax);
            const float scale = __expf(m_run - m_new);     // 0 on first chunk
            float sum = 0.0f;
#pragma unroll
            for (int k = 0; k < 8; ++k) {
                float e = __expf(sv[k] - m_new);
                sum += e;
                pbuf[srow * PLD + ssub + 8 * k] = (_Float16)e;
            }
#pragma unroll
            for (int msk = 1; msk < 8; msk <<= 1)
                sum += __shfl_xor(sum, msk, 32);
            l_run = l_run * scale + sum;
            m_run = m_new;
            if (ssub == 0) rsc[srow] = scale;
        }
        __syncthreads();

        // ---- Rescale accumulators, then O += P * V ----
        {
            float sc[8];
            const int rbase = rt * 16 + ((lane >> 4) << 3);
#pragma unroll
            for (int v = 0; v < 8; ++v) sc[v] = rsc[rbase + v];
#pragma unroll
            for (int t = 0; t < 12; ++t)
#pragma unroll
                for (int v = 0; v < 8; ++v) acc[t][v] *= sc[v];

            v16h a0 = frag_a(pbuf, PLD, rt * 16, 0,  lane);
            v16h a1 = frag_a(pbuf, PLD, rt * 16, 32, lane);
#pragma unroll
            for (int t = 0; t < 12; ++t) {                  // 24 WMMA
                const int dt = ct * 12 + t;
                v16h b0 = frag_b(kvT, MCP, dt * 16, 0,  lane);
                acc[t] = __builtin_amdgcn_wmma_f32_16x16x32_f16(
                        false, a0, false, b0, (short)0, acc[t], false, false);
                v16h b1 = frag_b(kvT, MCP, dt * 16, 32, lane);
                acc[t] = __builtin_amdgcn_wmma_f32_16x16x32_f16(
                        false, a1, false, b1, (short)0, acc[t], false, false);
            }
        }
    }

    // ---- Epilogue: normalize by 1/l and store fp32 ----
    __syncthreads();
    if ((tid & 7) == 0) rsc[tid >> 3] = 1.0f / l_run;
    __syncthreads();
    {
        float sc[8];
        const int rbase = rt * 16 + ((lane >> 4) << 3);
#pragma unroll
        for (int v = 0; v < 8; ++v) sc[v] = rsc[rbase + v];
#pragma unroll
        for (int t = 0; t < 12; ++t) {
            const int dt  = ct * 12 + t;
            const int col = dt * 16 + (lane & 15);
#pragma unroll
            for (int v = 0; v < 8; ++v) {
                const int n = n0 + rbase + v;
                out[((size_t)b * N_ + n) * D_ + col] = acc[t][v] * sc[v];
            }
        }
    }
}

extern "C" void kernel_launch(void* const* d_in, const int* in_sizes, int n_in,
                              void* d_out, int out_size, void* d_ws, size_t ws_size,
                              hipStream_t stream) {
    const float* hs  = (const float*)d_in[0];   // hidden_states  [B,N,D] fp32
    const float* img = (const float*)d_in[1];   // img_hidden_state [B,M,D] fp32
    float* out = (float*)d_out;

    dim3 grid(B_ * (N_ / BM));   // 1024 workgroups
    dim3 block(256);             // 8 wave32
    hipLaunchKernelGGL(ModalFusion_fa_kernel, grid, block, LDS_BYTES, stream,
                       hs, img, out);
}
